// Joint_model_3624952398162
// MI455X (gfx1250) — matrix operand order repaired and checked by reading, not compiled, not run on hardware
//
#include <hip/hip_runtime.h>

// ---------------------------------------------------------------------------
// Joint intent/slot model for MI455X (gfx1250, wave32, WMMA).
//
// All large GEMMs (~0.74 TFLOP total) run through v_wmma_f32_16x16x32_bf16.
// GEMM: 256x128 block tile, 8 waves each owning a 64x64 tile (4x4 WMMA tiles,
// 16 v_wmma per K=32 step against 16 ds_read_b128), LDS double-buffered so
// global loads of tile k+1 overlap the WMMA burst on tile k with one barrier
// per K-step. Weights are converted once per launch to bf16 in [N,K]
// (pre-transposed) layout so fragments are two aligned ds_read_b128 each,
// matching the CDNA5 16-bit WMMA VGPR layouts. Attention (62x62 per head,
// <2% of FLOPs) is a fused VALU kernel with K/V staged in LDS as bf16.
// Bias / residual / ReLU / bf16-output are fused into GEMM epilogues.
// ---------------------------------------------------------------------------

typedef __attribute__((ext_vector_type(16))) __bf16 bf16x16;
typedef __attribute__((ext_vector_type(8)))  float  v8f;

struct __attribute__((aligned(16))) U4 { unsigned int x, y, z, w; };
union FragU { U4 q[2]; bf16x16 f; };

__device__ __forceinline__ unsigned short f2bf(float f) {
  union { float f; unsigned int u; } v; v.f = f;
  unsigned int u = v.u;
  u += 0x7fffu + ((u >> 16) & 1u);      // round-to-nearest-even
  return (unsigned short)(u >> 16);
}
__device__ __forceinline__ float bf2f(unsigned short h) {
  union { unsigned int u; float f; } v; v.u = ((unsigned int)h) << 16;
  return v.f;
}

// ------------------------------ GEMM (WMMA) --------------------------------
// C[M,N] = A[M,K] * B[K,N], A bf16 row-major, B supplied pre-transposed as
// Bt[N,K] bf16. Output f32 and/or bf16 with fused bias/residual/relu.
#define GTM 256
#define GTN 128
#define GTK 32
#define LDA_S 20   // uints per LDS row (16 used + 4 pad -> 80B rows, 16B aligned)
#define LDB_S 20

enum { GF_BIAS = 1, GF_RES = 2, GF_RELU = 4, GF_BF16 = 8, GF_F32 = 16 };

__device__ __forceinline__ bf16x16 frag_a(const unsigned int* s, int rowBase, int lane) {
  // 16-bit A 16x32: lanes 0-15 K={0..7,16..23}, lanes 16-31 K={8..15,24..31}
  int row = rowBase + (lane & 15);
  int o = row * LDA_S + ((lane >> 4) << 2);
  FragU u;
  u.q[0] = *(const U4*)(s + o);
  u.q[1] = *(const U4*)(s + o + 8);
  return u.f;
}
__device__ __forceinline__ bf16x16 frag_b(const unsigned int* s, int rowBase, int lane) {
  // 16-bit B 32x16: lanes 0-15 K={0..7,16..23}, lanes 16-31 K={8..15,24..31} (n fixed per lane)
  int row = rowBase + (lane & 15);
  int o = row * LDB_S + ((lane >> 4) << 2);
  FragU u;
  u.q[0] = *(const U4*)(s + o);
  u.q[1] = *(const U4*)(s + o + 8);
  return u.f;
}

__global__ __launch_bounds__(256)
void gemm_bf16_kernel(const unsigned short* __restrict__ A,
                      const unsigned short* __restrict__ Bt,
                      const float* __restrict__ bias,
                      const float* __restrict__ res,
                      float* __restrict__ C,
                      unsigned short* __restrict__ Cb,
                      int M, int N, int K, int flags)
{
  __shared__ unsigned int sA[2][GTM * LDA_S];   // 2 x 20 KB
  __shared__ unsigned int sB[2][GTN * LDB_S];   // 2 x 10 KB

  const int tid  = threadIdx.x;
  const int lane = tid & 31;
  const int wid  = tid >> 5;
  const int wmq  = wid & 3;       // 4 wave-rows  -> 64 M each
  const int wnq  = wid >> 2;      // 2 wave-cols  -> 64 N each
  const int m0   = blockIdx.y * GTM;
  const int n0   = blockIdx.x * GTN;
  const int K2   = K >> 1;        // uints per row (K always multiple of 32)

  // A staging: one full 16-uint row per thread (rows 0..255)
  const int arow = tid;
  // B staging: half row per thread (rows 0..127, halves 0/8)
  const int brow  = tid >> 1;
  const int bhalf = (tid & 1) << 3;

  v8f acc[4][4];
  #pragma unroll
  for (int i = 0; i < 4; i++)
    #pragma unroll
    for (int j = 0; j < 4; j++) {
      v8f z = {};
      acc[i][j] = z;
    }

  U4 ra[4], rb[2];
  const U4 zz = {0, 0, 0, 0};

  // ---- prologue: fetch tile 0 into buffer 0 ----
  {
    ra[0] = ra[1] = ra[2] = ra[3] = zz;
    const int gr = m0 + arow;
    if (gr < M) {
      const unsigned int* g = (const unsigned int*)A + (long)gr * K2;
      ra[0] = *(const U4*)(g);
      ra[1] = *(const U4*)(g + 4);
      ra[2] = *(const U4*)(g + 8);
      ra[3] = *(const U4*)(g + 12);
    }
    rb[0] = rb[1] = zz;
    const int gn = n0 + brow;
    if (gn < N) {
      const unsigned int* g = (const unsigned int*)Bt + (long)gn * K2 + bhalf;
      rb[0] = *(const U4*)(g);
      rb[1] = *(const U4*)(g + 4);
    }
    unsigned int* da = sA[0] + arow * LDA_S;
    *(U4*)(da) = ra[0]; *(U4*)(da + 4) = ra[1];
    *(U4*)(da + 8) = ra[2]; *(U4*)(da + 12) = ra[3];
    unsigned int* db = sB[0] + brow * LDB_S + bhalf;
    *(U4*)(db) = rb[0]; *(U4*)(db + 4) = rb[1];
  }
  __syncthreads();

  int p = 0;
  for (int k0 = 0; k0 < K; k0 += GTK) {
    const bool more = (k0 + GTK < K);

    // ---- issue global loads for tile k+1 (overlap with WMMA burst) ----
    if (more) {
      ra[0] = ra[1] = ra[2] = ra[3] = zz;
      const int gr = m0 + arow;
      if (gr < M) {
        const unsigned int* g = (const unsigned int*)A + (long)gr * K2 + ((k0 + GTK) >> 1);
        ra[0] = *(const U4*)(g);
        ra[1] = *(const U4*)(g + 4);
        ra[2] = *(const U4*)(g + 8);
        ra[3] = *(const U4*)(g + 12);
      }
      rb[0] = rb[1] = zz;
      const int gn = n0 + brow;
      if (gn < N) {
        const unsigned int* g = (const unsigned int*)Bt + (long)gn * K2 + ((k0 + GTK) >> 1) + bhalf;
        rb[0] = *(const U4*)(g);
        rb[1] = *(const U4*)(g + 4);
      }
    }

    // ---- compute on buffer p: 16 v_wmma per K-step ----
    const unsigned int* cA = sA[p];
    const unsigned int* cB = sB[p];
    bf16x16 af[4], bfr[4];
    #pragma unroll
    for (int tm = 0; tm < 4; tm++) af[tm] = frag_a(cA, wmq * 64 + tm * 16, lane);
    #pragma unroll
    for (int tn = 0; tn < 4; tn++) bfr[tn] = frag_b(cB, wnq * 64 + tn * 16, lane);

    #pragma unroll
    for (int tm = 0; tm < 4; tm++)
      #pragma unroll
      for (int tn = 0; tn < 4; tn++)
        acc[tm][tn] = __builtin_amdgcn_wmma_f32_16x16x32_bf16(
            false, af[tm], false, bfr[tn], (short)0, acc[tm][tn], false, false);

    // ---- stage tile k+1 into the other buffer, single barrier ----
    if (more) {
      unsigned int* da = sA[1 - p] + arow * LDA_S;
      *(U4*)(da) = ra[0]; *(U4*)(da + 4) = ra[1];
      *(U4*)(da + 8) = ra[2]; *(U4*)(da + 12) = ra[3];
      unsigned int* db = sB[1 - p] + brow * LDB_S + bhalf;
      *(U4*)(db) = rb[0]; *(U4*)(db + 4) = rb[1];
      __syncthreads();
      p ^= 1;
    }
  }

  // Epilogue. C/D layout: VGPR e -> M = e (lanes 0-15), e+8 (lanes 16-31); N = lane&15.
  #pragma unroll
  for (int tm = 0; tm < 4; tm++) {
    const int rbase = m0 + wmq * 64 + tm * 16 + ((lane >> 4) << 3);
    #pragma unroll
    for (int tn = 0; tn < 4; tn++) {
      const int c = n0 + wnq * 64 + tn * 16 + (lane & 15);
      if (c < N) {
        const float bval = (flags & GF_BIAS) ? bias[c] : 0.0f;
        #pragma unroll
        for (int e = 0; e < 8; e++) {
          const int r = rbase + e;
          if (r < M) {
            float v = acc[tm][tn][e] + bval;
            if (flags & GF_RES)  v += res[(long)r * N + c];
            if (flags & GF_RELU) v = v > 0.0f ? v : 0.0f;
            if (flags & GF_F32)  C[(long)r * N + c]  = v;
            if (flags & GF_BF16) Cb[(long)r * N + c] = f2bf(v);
          }
        }
      }
    }
  }
}

// ------------------------- fused cross-attention ---------------------------
// One block per (batch, head). Q,K: [M,1536] bf16, V: [M,768] bf16.
// ctx (merged heads): [M,768] bf16.
__global__ __launch_bounds__(256)
void attn_kernel(const unsigned short* __restrict__ Q,
                 const unsigned short* __restrict__ Kt,
                 const unsigned short* __restrict__ V,
                 const int* __restrict__ mask,
                 unsigned short* __restrict__ ctx)
{
  __shared__ unsigned short sK[62 * 192];
  __shared__ unsigned short sV[62 * 96];
  __shared__ float sS[4 * 64];
  __shared__ float sMB[64];

  const int b = blockIdx.x >> 3;
  const int h = blockIdx.x & 7;
  const int tid = threadIdx.x;

  for (int i = tid; i < 62 * 192; i += 256) {
    int l = i / 192, d = i % 192;
    sK[i] = Kt[((long)(b * 62 + l)) * 1536 + h * 192 + d];
  }
  for (int i = tid; i < 62 * 96; i += 256) {
    int l = i / 96, d = i % 96;
    sV[i] = V[((long)(b * 62 + l)) * 768 + h * 96 + d];
  }
  if (tid < 64)
    sMB[tid] = (tid < 62) ? (1.0f - (float)mask[b * 62 + tid]) * -10000.0f : -3.0e38f;
  __syncthreads();

  const float scale = 0.07216878364870323f;   // 1/sqrt(192)

  for (int q0 = 0; q0 < 62; q0 += 4) {
    const int qi = tid >> 6;
    const int kk = tid & 63;
    const int q  = q0 + qi;
    float sc = -3.0e38f;
    if (q < 62 && kk < 62) {
      const unsigned short* qp = Q + ((long)(b * 62 + q)) * 1536 + h * 192;
      float a = 0.0f;
      for (int d = 0; d < 192; d++) a += bf2f(qp[d]) * bf2f(sK[kk * 192 + d]);
      sc = a * scale + sMB[kk];
    }
    sS[qi * 64 + kk] = sc;
    __syncthreads();

    if (tid < 4) {
      float mx = -3.4e38f;
      for (int k2 = 0; k2 < 64; k2++) mx = fmaxf(mx, sS[tid * 64 + k2]);
      float sum = 0.0f;
      for (int k2 = 0; k2 < 64; k2++) {
        float e = __expf(sS[tid * 64 + k2] - mx);
        sS[tid * 64 + k2] = e;
        sum += e;
      }
      const float inv = 1.0f / sum;
      for (int k2 = 0; k2 < 64; k2++) sS[tid * 64 + k2] *= inv;
    }
    __syncthreads();

    for (int idx = tid; idx < 4 * 96; idx += 256) {
      const int row = idx / 96, d = idx % 96;
      const int qq = q0 + row;
      if (qq < 62) {
        float a = 0.0f;
        for (int k2 = 0; k2 < 62; k2++) a += sS[row * 64 + k2] * bf2f(sV[k2 * 96 + d]);
        ctx[((long)(b * 62 + qq)) * 768 + h * 96 + d] = f2bf(a);
      }
    }
    __syncthreads();
  }
}

// ------------------------------ LayerNorm ----------------------------------
// One block per row of 768. out = g*(x[+res]-u)/sqrt(var+1e-12)+b, f32 + bf16.
__global__ __launch_bounds__(256)
void ln_kernel(const float* __restrict__ x, const float* __restrict__ res,
               const float* __restrict__ g, const float* __restrict__ bta,
               float* __restrict__ oF, unsigned short* __restrict__ oB)
{
  __shared__ float rb[256];
  const long r = blockIdx.x;
  const int tid = threadIdx.x;
  const float* xr = x + r * 768;
  const float* rr = res ? res + r * 768 : nullptr;

  float v[3];
  float s = 0.0f;
  #pragma unroll
  for (int j = 0; j < 3; j++) {
    const int i = tid + j * 256;
    v[j] = xr[i] + (rr ? rr[i] : 0.0f);
    s += v[j];
  }
  rb[tid] = s; __syncthreads();
  for (int o = 128; o > 0; o >>= 1) { if (tid < o) rb[tid] += rb[tid + o]; __syncthreads(); }
  const float u = rb[0] * (1.0f / 768.0f);
  __syncthreads();

  float sv = 0.0f;
  #pragma unroll
  for (int j = 0; j < 3; j++) { const float d = v[j] - u; sv += d * d; }
  rb[tid] = sv; __syncthreads();
  for (int o = 128; o > 0; o >>= 1) { if (tid < o) rb[tid] += rb[tid + o]; __syncthreads(); }
  const float inv = rsqrtf(rb[0] * (1.0f / 768.0f) + 1e-12f);

  #pragma unroll
  for (int j = 0; j < 3; j++) {
    const int i = tid + j * 256;
    const float o = g[i] * (v[j] - u) * inv + bta[i];
    oF[r * 768 + i] = o;
    oB[r * 768 + i] = f2bf(o);
  }
}

// --------------------------- small elementwise -----------------------------
__global__ void softmax_rows(const float* __restrict__ X, unsigned short* __restrict__ P,
                             int R, int C)
{
  const int r = blockIdx.x * 256 + threadIdx.x;
  if (r >= R) return;
  const float* x = X + (long)r * C;
  float mx = -3.4e38f;
  for (int c = 0; c < C; c++) mx = fmaxf(mx, x[c]);
  float s = 0.0f;
  for (int c = 0; c < C; c++) s += __expf(x[c] - mx);
  const float inv = 1.0f / s;
  for (int c = 0; c < C; c++) P[(long)r * C + c] = f2bf(__expf(x[c] - mx) * inv);
}

__global__ void cvt_kernel(const float* __restrict__ in, unsigned short* __restrict__ out, long n)
{
  const long i = (long)blockIdx.x * 256 + threadIdx.x;
  if (i < n) out[i] = f2bf(in[i]);
}

// out[n*K+k] = in[k*N+n] (bf16): pre-transpose weights to [N,K].
__global__ void cvt_t_kernel(const float* __restrict__ in, unsigned short* __restrict__ out,
                             int K, int N)
{
  const long i = (long)blockIdx.x * 256 + threadIdx.x;
  if (i >= (long)K * N) return;
  const int n = (int)(i / K);
  const int k = (int)(i % K);
  out[i] = f2bf(in[(long)k * N + n]);
}

// x = concat([h, h_left, h_right]) with h = concat([HI,HS]); bf16 out [M,4608].
__global__ void build_x_kernel(const float* __restrict__ HI, const float* __restrict__ HS,
                               unsigned short* __restrict__ X)
{
  const long idx = (long)blockIdx.x * 256 + threadIdx.x;
  if (idx >= (long)15872 * 4608) return;
  const int bl = (int)(idx / 4608);
  const int c  = (int)(idx % 4608);
  const int b = bl / 62, l = bl % 62;
  const int seg = c / 1536;
  const int cc  = c % 1536;
  const int sl = l + (seg == 1 ? -1 : (seg == 2 ? 1 : 0));
  float v = 0.0f;
  if (sl >= 0 && sl < 62) {
    const float* src = (cc < 768) ? HI : HS;
    v = src[((long)(b * 62 + sl)) * 768 + (cc & 767)];
  }
  X[idx] = f2bf(v);
}

__global__ void maxpool_add(const float* __restrict__ HI, const float* __restrict__ H,
                            float* __restrict__ out)
{
  const int idx = blockIdx.x * 256 + threadIdx.x;
  if (idx >= 256 * 768) return;
  const int b = idx / 768, d = idx % 768;
  float m = -3.4e38f;
  for (int l = 0; l < 62; l++) {
    const long o = ((long)(b * 62 + l)) * 768 + d;
    m = fmaxf(m, HI[o] + H[o]);
  }
  out[idx] = m;
}

__global__ void add_out(const float* __restrict__ HS, const float* __restrict__ H,
                        float* __restrict__ out)
{
  const long idx = (long)blockIdx.x * 256 + threadIdx.x;
  if (idx < (long)15872 * 768) out[idx] = HS[idx] + H[idx];
}

// ------------------------------- launcher ----------------------------------
static inline void launch_gemm(const unsigned short* A, const unsigned short* Bt,
                               const float* bias, const float* res,
                               float* C, unsigned short* Cb,
                               int M, int N, int K, int flags, hipStream_t s)
{
  dim3 g((N + GTN - 1) / GTN, (M + GTM - 1) / GTM);
  gemm_bf16_kernel<<<g, 256, 0, s>>>(A, Bt, bias, res, C, Cb, M, N, K, flags);
}

extern "C" void kernel_launch(void* const* d_in, const int* in_sizes, int n_in,
                              void* d_out, int out_size, void* d_ws, size_t ws_size,
                              hipStream_t stream)
{
  const int Mr = 256 * 62;        // 15872 rows
  const int D = 768, D2 = 1536, D6 = 4608, NB = 2;

  const float* Hin   = (const float*)d_in[0];
  const int*   maskp = (const int*)d_in[1];
  const float* Wi    = (const float*)d_in[2];   // [32,768]
  const float* Ws    = (const float*)d_in[3];   // [128,768]
  const float* Wq    = (const float*)d_in[4];
  const float* Bq    = (const float*)d_in[5];
  const float* Wk    = (const float*)d_in[6];
  const float* Bk    = (const float*)d_in[7];
  const float* Wv    = (const float*)d_in[8];
  const float* Bv    = (const float*)d_in[9];
  const float* Wqs   = (const float*)d_in[10];
  const float* Bqs   = (const float*)d_in[11];
  const float* Wks   = (const float*)d_in[12];
  const float* Bks   = (const float*)d_in[13];
  const float* Wvs   = (const float*)d_in[14];
  const float* Bvs   = (const float*)d_in[15];
  const float* WoI   = (const float*)d_in[16];
  const float* BoI   = (const float*)d_in[17];
  const float* lnIw  = (const float*)d_in[18];
  const float* lnIb  = (const float*)d_in[19];
  const float* WoS   = (const float*)d_in[20];
  const float* BoS   = (const float*)d_in[21];
  const float* lnSw  = (const float*)d_in[22];
  const float* lnSb  = (const float*)d_in[23];
  const float* Win   = (const float*)d_in[24];
  const float* Bin   = (const float*)d_in[25];
  const float* Wout  = (const float*)d_in[26];
  const float* Bout  = (const float*)d_in[27];
  const float* lnFIw = (const float*)d_in[28];
  const float* lnFIb = (const float*)d_in[29];
  const float* lnFSw = (const float*)d_in[30];
  const float* lnFSb = (const float*)d_in[31];

  // bump allocator over d_ws (≈720 MB total; ws assumed sized by harness)
  size_t off = 0;
  auto alloc = [&](size_t elems, size_t esz) -> void* {
    off = (off + 255) & ~(size_t)255;
    void* p = (char*)d_ws + off;
    off += elems * esz;
    return p;
  };
  typedef unsigned short us;
  us* WqT   = (us*)alloc((size_t)NB * D2 * D, 2);
  us* WkT   = (us*)alloc((size_t)NB * D2 * D, 2);
  us* WqsT  = (us*)alloc((size_t)NB * D2 * D, 2);
  us* WksT  = (us*)alloc((size_t)NB * D2 * D, 2);
  us* WvT   = (us*)alloc((size_t)NB * D * D, 2);
  us* WvsT  = (us*)alloc((size_t)NB * D * D, 2);
  us* WoIT  = (us*)alloc((size_t)NB * D * D, 2);
  us* WoST  = (us*)alloc((size_t)NB * D * D, 2);
  us* WoutT = (us*)alloc((size_t)NB * D * D, 2);
  us* WinT  = (us*)alloc((size_t)NB * D * D6, 2);
  us* WiBt  = (us*)alloc((size_t)32 * D, 2);    // = Wi (already [N=32,K=768])
  us* WiT   = (us*)alloc((size_t)D * 32, 2);    // Wi transposed -> [768,32]
  us* WsBt  = (us*)alloc((size_t)128 * D, 2);
  us* WsT   = (us*)alloc((size_t)D * 128, 2);
  us* Hbf   = (us*)alloc((size_t)Mr * D, 2);
  float* HIinF = (float*)alloc((size_t)Mr * D, 4);
  us*    HIinB = (us*)  alloc((size_t)Mr * D, 2);
  float* HSinF = (float*)alloc((size_t)Mr * D, 4);
  us*    HSinB = (us*)  alloc((size_t)Mr * D, 2);
  us* bufQ = (us*)alloc((size_t)Mr * D2, 2);
  us* bufK = (us*)alloc((size_t)Mr * D2, 2);
  us* bufV = (us*)alloc((size_t)Mr * D, 2);     // also reused as bf16 hid1
  us* ctxB = (us*)alloc((size_t)Mr * D, 2);
  float* tbuf = (float*)alloc((size_t)Mr * D, 4);
  float* HI1F = (float*)alloc((size_t)Mr * D, 4);
  us*    HI1B = (us*)  alloc((size_t)Mr * D, 2);
  float* HS1F = (float*)alloc((size_t)Mr * D, 4);
  us*    HS1B = (us*)  alloc((size_t)Mr * D, 2);
  us* Xb  = (us*)alloc((size_t)Mr * D6, 2);
  float* lg = (float*)alloc((size_t)Mr * 128, 4);
  us* Pb  = (us*)alloc((size_t)Mr * 128, 2);
  (void)ws_size; (void)in_sizes; (void)n_in; (void)out_size;

  auto cvt = [&](const float* in, us* out, long n) {
    cvt_kernel<<<(unsigned)((n + 255) / 256), 256, 0, stream>>>(in, out, n);
  };
  auto cvtT = [&](const float* in, us* out, int K, int N) {
    long n = (long)K * N;
    cvt_t_kernel<<<(unsigned)((n + 255) / 256), 256, 0, stream>>>(in, out, K, N);
  };

  // ---- one-time weight conversion (bf16, pre-transposed to [N,K]) ----
  for (int i = 0; i < NB; i++) {
    cvtT(Wq  + (size_t)i * D * D2, WqT  + (size_t)i * D2 * D, D, D2);
    cvtT(Wk  + (size_t)i * D * D2, WkT  + (size_t)i * D2 * D, D, D2);
    cvtT(Wqs + (size_t)i * D * D2, WqsT + (size_t)i * D2 * D, D, D2);
    cvtT(Wks + (size_t)i * D * D2, WksT + (size_t)i * D2 * D, D, D2);
    cvtT(Wv  + (size_t)i * D * D,  WvT  + (size_t)i * D * D,  D, D);
    cvtT(Wvs + (size_t)i * D * D,  WvsT + (size_t)i * D * D,  D, D);
    cvtT(WoI + (size_t)i * D * D,  WoIT + (size_t)i * D * D,  D, D);
    cvtT(WoS + (size_t)i * D * D,  WoST + (size_t)i * D * D,  D, D);
    cvtT(Wout+ (size_t)i * D * D,  WoutT+ (size_t)i * D * D,  D, D);
    cvtT(Win + (size_t)i * D6 * D, WinT + (size_t)i * D * D6, D6, D);
  }
  cvt(Wi, WiBt, (long)32 * D);
  cvtT(Wi, WiT, 32, D);
  cvt(Ws, WsBt, (long)128 * D);
  cvtT(Ws, WsT, 128, D);
  cvt(Hin, Hbf, (long)Mr * D);

  // label attention: out = softmax(A @ W^T) @ W + res
  auto run_label = [&](const us* Abf, const us* WBt, const us* WT, int C,
                       const float* resF, float* outF, us* outB) {
    launch_gemm(Abf, WBt, nullptr, nullptr, lg, nullptr, Mr, C, D, GF_F32, stream);
    softmax_rows<<<(Mr + 255) / 256, 256, 0, stream>>>(lg, Pb, Mr, C);
    launch_gemm(Pb, WT, nullptr, resF, outF, outB, Mr, D, C,
                GF_RES | GF_F32 | GF_BF16, stream);
  };

  auto run_block = [&](int i) {
    const size_t o2 = (size_t)i * D2 * D, o1 = (size_t)i * D * D;
    // intent branch: Q from HI, K/V from HS
    launch_gemm(HIinB, WqT + o2, Bq + (size_t)i * D2, nullptr, nullptr, bufQ,
                Mr, D2, D, GF_BIAS | GF_BF16, stream);
    launch_gemm(HSinB, WkT + o2, Bk + (size_t)i * D2, nullptr, nullptr, bufK,
                Mr, D2, D, GF_BIAS | GF_BF16, stream);
    launch_gemm(HSinB, WvT + o1, Bv + (size_t)i * D, nullptr, nullptr, bufV,
                Mr, D, D, GF_BIAS | GF_BF16, stream);
    attn_kernel<<<256 * 8, 256, 0, stream>>>(bufQ, bufK, bufV, maskp, ctxB);
    launch_gemm(ctxB, WoIT + o1, BoI + (size_t)i * D, HIinF, tbuf, nullptr,
                Mr, D, D, GF_BIAS | GF_RES | GF_F32, stream);
    ln_kernel<<<Mr, 256, 0, stream>>>(tbuf, nullptr, lnIw + (size_t)i * D,
                                      lnIb + (size_t)i * D, HI1F, HI1B);
    // slot branch: Q from HS, K/V from HI
    launch_gemm(HSinB, WqsT + o2, Bqs + (size_t)i * D2, nullptr, nullptr, bufQ,
                Mr, D2, D, GF_BIAS | GF_BF16, stream);
    launch_gemm(HIinB, WksT + o2, Bks + (size_t)i * D2, nullptr, nullptr, bufK,
                Mr, D2, D, GF_BIAS | GF_BF16, stream);
    launch_gemm(HIinB, WvsT + o1, Bvs + (size_t)i * D, nullptr, nullptr, bufV,
                Mr, D, D, GF_BIAS | GF_BF16, stream);
    attn_kernel<<<256 * 8, 256, 0, stream>>>(bufQ, bufK, bufV, maskp, ctxB);
    launch_gemm(ctxB, WoST + o1, BoS + (size_t)i * D, HSinF, tbuf, nullptr,
                Mr, D, D, GF_BIAS | GF_RES | GF_F32, stream);
    ln_kernel<<<Mr, 256, 0, stream>>>(tbuf, nullptr, lnSw + (size_t)i * D,
                                      lnSb + (size_t)i * D, HS1F, HS1B);
    // neighbor-context FFN
    {
      long n = (long)Mr * D6;
      build_x_kernel<<<(unsigned)((n + 255) / 256), 256, 0, stream>>>(HI1F, HS1F, Xb);
    }
    launch_gemm(Xb, WinT + (size_t)i * D * D6, Bin + (size_t)i * D, nullptr,
                nullptr, bufV, Mr, D, D6, GF_BIAS | GF_RELU | GF_BF16, stream);
    launch_gemm(bufV, WoutT + o1, Bout + (size_t)i * D, nullptr, tbuf, nullptr,
                Mr, D, D, GF_BIAS | GF_F32, stream);
    ln_kernel<<<Mr, 256, 0, stream>>>(tbuf, HI1F, lnFIw + (size_t)i * D,
                                      lnFIb + (size_t)i * D, HIinF, HIinB);
    ln_kernel<<<Mr, 256, 0, stream>>>(tbuf, HS1F, lnFSw + (size_t)i * D,
                                      lnFSb + (size_t)i * D, HSinF, HSinB);
  };

  // ---- forward pass ----
  run_label(Hbf, WiBt, WiT, 32, Hin, HIinF, HIinB);     // H_I = la(H) + H
  run_label(Hbf, WsBt, WsT, 128, Hin, HSinF, HSinB);    // H_S = la(H) + H
  run_block(0);
  run_label(HIinB, WiBt, WiT, 32, HIinF, HIinF, HIinB); // H_I += la(H_I)
  run_label(HSinB, WsBt, WsT, 128, HSinF, HSinF, HSinB);
  run_block(1);

  float* out = (float*)d_out;
  maxpool_add<<<(256 * 768 + 255) / 256, 256, 0, stream>>>(HIinF, Hin, out);
  {
    long n = (long)Mr * D;
    add_out<<<(unsigned)((n + 255) / 256), 256, 0, stream>>>(HSinF, Hin, out + 256 * 768);
  }
}
